// FixedChannelDP_83047487635647
// MI455X (gfx1250) — matrix operand order repaired and checked by reading, not compiled, not run on hardware
//
#include <hip/hip_runtime.h>

// Dual-pol complex FIR ('full' convolution, L=513 taps) via bf16 WMMA with
// hi/lo split (bf16x3) for ~fp32 accuracy, fp32 accumulation.
//
//   y[j] = sum_t h[t] * x[j - t],  j in [0, N+L-1)
//
// Tile: 256 outputs per wave; A[m][k] = x[base+16m+k+1-32c] (16x32 bf16),
// B_c[k][n] = h[32c+n-k-1] (32x16 Toeplitz), 18 chunks cover all 513 taps.
// All 18 chunks' Toeplitz tiles are built ONCE per block into LDS, so the
// main loop is barrier-free: pure ds_load_b128 + v_wmma.

#define N_SAMP   4194304
#define L_TAPS   513
#define N_OUT    (N_SAMP + L_TAPS - 1)   // 4194816 = 256 * 16386 exactly
#define TILE     256
#define N_TILES  (N_OUT / TILE)          // 16386
#define CHUNKS   18
#define S_LEN    1632                    // staged x window (bf16 halves)
#define TPB      256                     // 8 wave32s
#define TPB_TILES 4                      // tile positions per block (x2 pols)

typedef __attribute__((ext_vector_type(16))) __bf16 v16bf;
typedef __attribute__((ext_vector_type(8)))  float  v8f;

union V16U { uint4 q[2]; v16bf v; };

__device__ __forceinline__ unsigned short bf_hi(float f) {
  return (unsigned short)(__float_as_uint(f) >> 16);   // truncate to bf16
}
__device__ __forceinline__ float bf_f(unsigned short b) {
  return __uint_as_float(((unsigned)b) << 16);
}

#define WMMA_BF16(A, B, C) \
  __builtin_amdgcn_wmma_f32_16x16x32_bf16(false, (A), false, (B), (short)0, (C), false, false)

__global__ __launch_bounds__(TPB) void fixed_channel_dp_wmma(
    const float* __restrict__ txr, const float* __restrict__ txi,
    const float* __restrict__ hre, const float* __restrict__ him,
    float2* __restrict__ out)
{
  // x windows: [pol*4 + {r_hi, r_lo, i_hi, i_lo}][s]          (26112 B)
  __shared__ __align__(16) unsigned short sx[8][S_LEN];
  // ALL chunks' Toeplitz taps, per-lane swizzled:
  // [chunk][variant][lane][16]                                 (73728 B)
  __shared__ __align__(32) unsigned short sB[CHUNKS][4][32][16];
  // taps hi/lo: {hr_hi, hr_lo, hi_hi, hi_lo}                   (4160 B)
  __shared__ __align__(16) unsigned short sh[4][520];

  const int tid  = threadIdx.x;
  const int lane = tid & 31;
  const int wave = tid >> 5;
  const int pol  = wave >> 2;

  const int tile0 = blockIdx.x * TPB_TILES;
  const int B0    = tile0 * TILE;

  // ---- stage x window (both pols, hi/lo bf16 split), zero-padded ----
  for (int s = tid; s < S_LEN; s += TPB) {
    const int gx = B0 - 575 + s;
    const bool ok = (gx >= 0) && (gx < N_SAMP);
    const int g = ok ? gx : 0;
    const float r0 = ok ? txr[g] : 0.f;
    const float i0 = ok ? txi[g] : 0.f;
    const float r1 = ok ? txr[N_SAMP + g] : 0.f;
    const float i1 = ok ? txi[N_SAMP + g] : 0.f;
    unsigned short h;
    h = bf_hi(r0); sx[0][s] = h; sx[1][s] = bf_hi(r0 - bf_f(h));
    h = bf_hi(i0); sx[2][s] = h; sx[3][s] = bf_hi(i0 - bf_f(h));
    h = bf_hi(r1); sx[4][s] = h; sx[5][s] = bf_hi(r1 - bf_f(h));
    h = bf_hi(i1); sx[6][s] = h; sx[7][s] = bf_hi(i1 - bf_f(h));
  }
  // ---- stage filter taps hi/lo ----
  for (int t = tid; t < 520; t += TPB) {
    const float vr = (t < L_TAPS) ? hre[t] : 0.f;
    const float vi = (t < L_TAPS) ? him[t] : 0.f;
    unsigned short h;
    h = bf_hi(vr); sh[0][t] = h; sh[1][t] = bf_hi(vr - bf_f(h));
    h = bf_hi(vi); sh[2][t] = h; sh[3][t] = bf_hi(vi - bf_f(h));
  }
  __syncthreads();

  // ---- build ALL Toeplitz B tiles once (loop-invariant across tiles) ----
  // B layout: lane holds N=lane&15; K = (lane>=16 ? 8 : 0) + elem + (elem>=8 ? 8 : 0)
  for (int e = tid; e < CHUNKS * 2048; e += TPB) {
    const int c    = e >> 11;
    const int rem  = e & 2047;
    const int v4   = rem >> 9;
    const int rem2 = rem & 511;
    const int l    = rem2 >> 4;
    const int idx  = rem2 & 15;
    const int n    = l & 15;
    const int k    = ((l >> 4) << 3) + idx + ((idx >> 3) << 3);
    const int t    = 32 * c + n - k - 1;
    sB[c][v4][l][idx] = (t >= 0 && t < L_TAPS) ? sh[v4][t] : (unsigned short)0;
  }
  __syncthreads();

  int myTile = tile0 + (wave & 3);
  if (myTile > N_TILES - 1) myTile = N_TILES - 1;   // tail: duplicate tile, identical stores
  const int slot = myTile - tile0;                  // 0..3
  const int base = myTile * TILE;

  // 16-bit A layout: lane holds M=lane&15; K pairs per VGPR, +8 for upper lanes
  const int M    = lane & 15;
  const int hk   = (lane >> 4) << 3;                // 0 or 8
  const int sOff = slot * 256 + 576 + 16 * M + hk;  // multiple of 8 -> b128 aligned

  v8f accR = {0.f,0.f,0.f,0.f,0.f,0.f,0.f,0.f};     // conv(x_r, h_r) terms
  v8f accN = {0.f,0.f,0.f,0.f,0.f,0.f,0.f,0.f};     // conv(x_i, h_i) terms (subtracted)
  v8f accI = {0.f,0.f,0.f,0.f,0.f,0.f,0.f,0.f};     // conv(x_r,h_i) + conv(x_i,h_r)

  const unsigned short (*sxp)[S_LEN] = sx + pol * 4;

  // ---- barrier-free main loop: 16 ds_load_b128 + 12 WMMA per chunk ----
  for (int c = 0; c < CHUNKS; ++c) {
    const v16bf Brh = *(const v16bf*)&sB[c][0][lane][0];
    const v16bf Brl = *(const v16bf*)&sB[c][1][lane][0];
    const v16bf Bih = *(const v16bf*)&sB[c][2][lane][0];
    const v16bf Bil = *(const v16bf*)&sB[c][3][lane][0];

    const int sA = sOff - 32 * c;
    V16U Arh, Arl, Aih, Ail;
    Arh.q[0] = *(const uint4*)&sxp[0][sA]; Arh.q[1] = *(const uint4*)&sxp[0][sA + 16];
    Arl.q[0] = *(const uint4*)&sxp[1][sA]; Arl.q[1] = *(const uint4*)&sxp[1][sA + 16];
    Aih.q[0] = *(const uint4*)&sxp[2][sA]; Aih.q[1] = *(const uint4*)&sxp[2][sA + 16];
    Ail.q[0] = *(const uint4*)&sxp[3][sA]; Ail.q[1] = *(const uint4*)&sxp[3][sA + 16];

    // y_r (+) terms: x_r * h_r  (hi*hi + lo*hi + hi*lo)
    accR = WMMA_BF16(Arh.v, Brh, accR);
    accR = WMMA_BF16(Arl.v, Brh, accR);
    accR = WMMA_BF16(Arh.v, Brl, accR);
    // y_r (-) terms: x_i * h_i
    accN = WMMA_BF16(Aih.v, Bih, accN);
    accN = WMMA_BF16(Ail.v, Bih, accN);
    accN = WMMA_BF16(Aih.v, Bil, accN);
    // y_i terms: x_r * h_i + x_i * h_r
    accI = WMMA_BF16(Arh.v, Bih, accI);
    accI = WMMA_BF16(Arl.v, Bih, accI);
    accI = WMMA_BF16(Arh.v, Bil, accI);
    accI = WMMA_BF16(Aih.v, Brh, accI);
    accI = WMMA_BF16(Ail.v, Brh, accI);
    accI = WMMA_BF16(Aih.v, Brl, accI);
  }

  // C/D layout: lane holds N=lane&15; VGPR r -> M = r + 8*(lane>=16)
  const int nn  = lane & 15;
  const int mhi = (lane >> 4) << 3;
  float2* op = out + (size_t)pol * N_OUT + base + nn;
  #pragma unroll
  for (int r = 0; r < 8; ++r) {
    float2 v;
    v.x = accR[r] - accN[r];
    v.y = accI[r];
    op[16 * (r + mhi)] = v;   // coalesced b64 stores
  }
}

extern "C" void kernel_launch(void* const* d_in, const int* in_sizes, int n_in,
                              void* d_out, int out_size, void* d_ws, size_t ws_size,
                              hipStream_t stream) {
  const float* txr = (const float*)d_in[0];   // [2, N]
  const float* txi = (const float*)d_in[1];   // [2, N]
  const float* hre = (const float*)d_in[2];   // [L]
  const float* him = (const float*)d_in[3];   // [L]
  float2* out = (float2*)d_out;               // [2, N_OUT] of (re, im)

  (void)in_sizes; (void)n_in; (void)out_size; (void)d_ws; (void)ws_size;

  const int nblocks = (N_TILES + TPB_TILES - 1) / TPB_TILES;  // 4097
  fixed_channel_dp_wmma<<<dim3(nblocks), dim3(TPB), 0, stream>>>(txr, txi, hre, him, out);
}
